// MultiHeadAttention_79963701117481
// MI455X (gfx1250) — compile-verified, matrix operand order checked by reading, not run on hardware
//
#include <hip/hip_runtime.h>
#include <hip/hip_bf16.h>

typedef __attribute__((ext_vector_type(16))) __bf16 v16bf;
typedef __attribute__((ext_vector_type(8)))  float  v8f;
typedef int v4i_t __attribute__((vector_size(16)));

#define B_  2
#define S_  2048
#define D_  1024
#define H_  16
#define DH_ 64

#if defined(__HIP_DEVICE_COMPILE__) && defined(__gfx1250__) && \
    __has_builtin(__builtin_amdgcn_global_load_async_to_lds_b128)
#define HAVE_ASYNC_LDS 1
#endif

// ---------------------------------------------------------------------------
// Fragment-resident layouts: one WMMA operand = 32 lanes x 16 bf16 = 1KB,
// stored [lane][elem] contiguous -> each lane loads 32B (2x b128), a wave
// reads 1KB fully coalesced.
// 16-bit operand mapping (ISA 7.12.2): lane L: m|n=L%16, g=L/16;
//   k = e + 8g (e<8) ; k = e + 8 + 8g (e>=8)
//   inverse: g = (k>>3)&1 ; e = k - 8g - 8*(k>>4)
// C/D f32 tile: lane col = L%16, VGPR r -> row r + 8*(L/16).
// ---------------------------------------------------------------------------
__device__ __forceinline__ void inv_k(int k, int& g, int& e) {
  g = (k >> 3) & 1;
  e = k - 8 * g - 8 * (k >> 4);
}

__device__ __forceinline__ v16bf load_frag(const __bf16* base, int lane) {
  return *(const v16bf*)(base + lane * 16);
}

__device__ __forceinline__ v8f wmma_bf16(v16bf a, v16bf b, v8f c) {
  return __builtin_amdgcn_wmma_f32_16x16x32_bf16(false, a, false, b, (short)0, c,
                                                 false, false);
}

// ---------------------------------------------------------------------------
// Global -> LDS staging: async (GLOBAL_LOAD_ASYNC_TO_LDS_B128, ASYNCcnt) when
// available, else VGPR-bounced b128 copy. Compile-time chunk counts so the
// async issues are straight-line (no exec-mask loop).
// ---------------------------------------------------------------------------
template <int NCHUNKS, int STEP>
__device__ __forceinline__ void stage16(__bf16* lds, const __bf16* gsrc,
                                        int tid) {
  uint4* ld = (uint4*)lds;
  const uint4* gs = (const uint4*)gsrc;
#if defined(HAVE_ASYNC_LDS)
  typedef __attribute__((address_space(1))) void* as1vp;
  typedef __attribute__((address_space(3))) void* as3vp;
  typedef __attribute__((address_space(1))) v4i_t* as1p;
  typedef __attribute__((address_space(3))) v4i_t* as3p;
#pragma unroll
  for (int u = 0; u < NCHUNKS / STEP; ++u) {
    const int c = tid + u * STEP;
    __builtin_amdgcn_global_load_async_to_lds_b128(
        (as1p)(as1vp)(const_cast<uint4*>(gs) + c),
        (as3p)(as3vp)(ld + c), 0, 0);
  }
#else
#pragma unroll
  for (int u = 0; u < NCHUNKS / STEP; ++u) {
    const int c = tid + u * STEP;
    ld[c] = gs[c];
  }
#endif
}

template <int N>
__device__ __forceinline__ void async_wait() {
#if defined(HAVE_ASYNC_LDS)
#if __has_builtin(__builtin_amdgcn_s_wait_asynccnt)
  __builtin_amdgcn_s_wait_asynccnt(N);
#else
  asm volatile("s_wait_asynccnt %0" ::"i"(N) : "memory");
#endif
#endif
}

// ---------------------------------------------------------------------------
// Converters: f32 -> bf16 into fragment-tiled layouts.
// ---------------------------------------------------------------------------
// X[4096,1024] -> Xs[mTile(256)][kTile(32)][512]
__global__ __launch_bounds__(256) void conv_X(__bf16* __restrict__ dst,
                                              const float* __restrict__ src) {
  const int o = blockIdx.x * 256 + threadIdx.x;
  const int frag = o >> 9, w = o & 511;
  const int lane = w >> 4, e = w & 15;
  const int m = lane & 15, g = lane >> 4;
  const int k = e + 8 * g + (e >= 8 ? 8 : 0);
  const int mT = frag >> 5, kT = frag & 31;
  dst[o] = (__bf16)src[(size_t)(mT * 16 + m) * D_ + kT * 32 + k];
}

struct WcArgs {
  const float* src[3];
  __bf16* dst[3];
};

// W[h,d,e](H,D,DH) -> Ws[kTile(32)][nTile(64)][512], col = h*64+e
__global__ __launch_bounds__(256) void conv_Whead3(WcArgs args) {
  const int o = blockIdx.x * 256 + threadIdx.x;
  const float* src = args.src[blockIdx.y];
  __bf16* dst = args.dst[blockIdx.y];
  const int frag = o >> 9, w = o & 511;
  const int lane = w >> 4, e = w & 15;
  const int nl = lane & 15, g = lane >> 4;
  const int k = e + 8 * g + (e >= 8 ? 8 : 0);
  const int kT = frag >> 6, nT = frag & 63;
  const int d = kT * 32 + k, c = nT * 16 + nl;
  const int h = c >> 6, edh = c & 63;
  dst[o] = (__bf16)src[((size_t)h * D_ + d) * DH_ + edh];
}

// Wo[D,D] row-major -> Ws[kTile][nTile][512]
__global__ __launch_bounds__(256) void conv_Wo(__bf16* __restrict__ dst,
                                               const float* __restrict__ src) {
  const int o = blockIdx.x * 256 + threadIdx.x;
  const int frag = o >> 9, w = o & 511;
  const int lane = w >> 4, e = w & 15;
  const int nl = lane & 15, g = lane >> 4;
  const int k = e + 8 * g + (e >= 8 ? 8 : 0);
  const int kT = frag >> 6, nT = frag & 63;
  dst[o] = (__bf16)src[(size_t)(kT * 32 + k) * D_ + nT * 16 + nl];
}

// ---------------------------------------------------------------------------
// GEMM mainloop: (4096 x 1024) x (1024 x 1024), operands fragment-tiled.
// Block 256 thr = 8 waves; tile 128(M) x 128(N); wave = 32(M) x 64(N);
// double-buffered async B staging, 8 wmma / wave / K-step, 32 steps.
// ---------------------------------------------------------------------------
__device__ __forceinline__ void wmma_mainloop(const __bf16* __restrict__ X,
                                              const __bf16* __restrict__ W,
                                              __bf16* Bs /* 2 x 4096 */,
                                              int m0, int n0, v8f acc[2][4]) {
  const int tid = threadIdx.x;
  const int wave = tid >> 5, lane = tid & 31;
  const int wm = wave >> 1, wn = wave & 1;
  const int mT0 = (m0 >> 4) + wm * 2;
  const int nT0 = n0 >> 4;

  stage16<512, 256>(Bs, W + (size_t)nT0 * 512, tid);
  for (int it = 0; it < 32; ++it) {
    if (it + 1 < 32) {
      stage16<512, 256>(Bs + ((it + 1) & 1) * 4096,
                        W + ((size_t)(it + 1) * 64 + nT0) * 512, tid);
      async_wait<2>();  // 2 chunks/thread in flight for next stage
    } else {
      async_wait<0>();
    }
    __syncthreads();
    const __bf16* Bc = Bs + (it & 1) * 4096;
#pragma unroll
    for (int i = 0; i < 2; ++i) {
      if (it + 1 < 32)
        __builtin_prefetch(X + ((size_t)(mT0 + i) * 32 + it + 1) * 512 + lane * 16,
                           0, 3);
      const v16bf a = load_frag(X + ((size_t)(mT0 + i) * 32 + it) * 512, lane);
#pragma unroll
      for (int nf = 0; nf < 4; ++nf) {
        const v16bf b = load_frag(&Bc[(wn * 4 + nf) * 512], lane);
        acc[i][nf] = wmma_bf16(a, b, acc[i][nf]);
      }
    }
    __syncthreads();
  }
}

struct QKVArgs {
  const __bf16* W[3];
  const float* bias[3];
  __bf16* out[3];
};

// z = 0: Q frag store, 1: K^T frag store (same formula), 2: V frag store
__global__ __launch_bounds__(256) void gemm_qkv(const __bf16* __restrict__ X,
                                                QKVArgs args) {
  __shared__ __align__(32) __bf16 Bs[2 * 8 * 512];
  const int z = blockIdx.z;
  const __bf16* W = args.W[z];
  const float* bias = args.bias[z];
  __bf16* Outb = args.out[z];
  const int m0 = blockIdx.x * 128, n0 = blockIdx.y * 128;

  v8f acc[2][4];
  const v8f zero = {};
#pragma unroll
  for (int i = 0; i < 2; ++i)
#pragma unroll
    for (int nf = 0; nf < 4; ++nf) acc[i][nf] = zero;

  wmma_mainloop(X, W, Bs, m0, n0, acc);

  const int tid = threadIdx.x, wave = tid >> 5, lane = tid & 31;
  const int wm = wave >> 1, wn = wave & 1;
  const int nn = lane & 15, gc = lane >> 4;
  const bool isV = (z == 2);
#pragma unroll
  for (int i = 0; i < 2; ++i) {
#pragma unroll
    for (int nf = 0; nf < 4; ++nf) {
      const int col = n0 + (wn * 4 + nf) * 16 + nn;  // 0..1023
      const float bs = bias[col];
#pragma unroll
      for (int r = 0; r < 8; ++r) {
        const int gm = m0 + (wm * 2 + i) * 16 + r + 8 * gc;  // 0..4095
        const float val = acc[i][nf][r] + bs;
        const int b = gm >> 11, s = gm & (S_ - 1);
        const int h = col >> 6, edh = col & 63;
        const int bh = b * H_ + h;
        int g, e;
        size_t off;
        if (!isV) {  // Q / K^T: [bh][sT(128)][fT(2)][512], lane%16 = seq
          inv_k(edh & 31, g, e);
          off = (((size_t)bh * 128 + (s >> 4)) * 2 + (edh >> 5)) * 512 +
                ((s & 15) + 16 * g) * 16 + e;
        } else {     // V: [bh][tT(64)][eT(4)][512], k = seq
          inv_k(s & 31, g, e);
          off = (((size_t)bh * 64 + (s >> 5)) * 4 + (edh >> 4)) * 512 +
                ((edh & 15) + 16 * g) * 16 + e;
        }
        Outb[off] = (__bf16)val;
      }
    }
  }
}

__global__ __launch_bounds__(256) void gemm_out(const __bf16* __restrict__ X,
                                                const __bf16* __restrict__ W,
                                                const float* __restrict__ bias,
                                                float* __restrict__ Out) {
  __shared__ __align__(32) __bf16 Bs[2 * 8 * 512];
  const int m0 = blockIdx.x * 128, n0 = blockIdx.y * 128;

  v8f acc[2][4];
  const v8f zero = {};
#pragma unroll
  for (int i = 0; i < 2; ++i)
#pragma unroll
    for (int nf = 0; nf < 4; ++nf) acc[i][nf] = zero;

  wmma_mainloop(X, W, Bs, m0, n0, acc);

  const int tid = threadIdx.x, wave = tid >> 5, lane = tid & 31;
  const int wm = wave >> 1, wn = wave & 1;
  const int nn = lane & 15, gc = lane >> 4;
#pragma unroll
  for (int i = 0; i < 2; ++i) {
#pragma unroll
    for (int nf = 0; nf < 4; ++nf) {
      const int col = n0 + (wn * 4 + nf) * 16 + nn;
      const float bs = bias[col];
#pragma unroll
      for (int r = 0; r < 8; ++r) {
        const int gm = m0 + (wm * 2 + i) * 16 + r + 8 * gc;
        Out[(size_t)gm * D_ + col] = acc[i][nf][r] + bs;
      }
    }
  }
}

// ---------------------------------------------------------------------------
// Flash attention: grid (S/64, B*H), block 128 (4 waves, 16 q-rows each).
// 64 key columns per iteration: 8 score wmma + 8 ctx wmma; double-buffered
// async K/V staging; online softmax with 16-lane shuffle row reductions.
// Qf: [bh][sT(128)][kT(2)][512]  Kf: [bh][tT(128)][eT(2)][512]
// Vf: [bh][tT(64)][eT(4)][512]   Cf (A-frag like Xs): [mT(256)][kT(32)][512]
// ---------------------------------------------------------------------------
__global__ __launch_bounds__(128) void attn(const __bf16* __restrict__ Qf,
                                            const __bf16* __restrict__ Kf,
                                            const __bf16* __restrict__ Vf,
                                            __bf16* __restrict__ Cf) {
  __shared__ __align__(32) __bf16 Ks[2 * 8 * 512];  // 2 x (4 tT x 2 eT)
  __shared__ __align__(32) __bf16 Vs[2 * 8 * 512];  // 2 x (2 tT x 4 eT)
  __shared__ __align__(32) __bf16 Ps[4][2 * 512];   // per-wave probs (A frags)

  const int bh = blockIdx.y;
  const int s0 = blockIdx.x * 64;
  const int tid = threadIdx.x;
  const int wave = tid >> 5, lane = tid & 31;
  const int nn = lane & 15, gc = lane >> 4;

  const size_t qbase = ((size_t)bh * 128 + (s0 >> 4) + wave) * 2;
  const v16bf qa0 = load_frag(Qf + qbase * 512, lane);
  const v16bf qa1 = load_frag(Qf + (qbase + 1) * 512, lane);

  float mrow[8], lrow[8];
  v8f o[4];
  const v8f zero = {};
#pragma unroll
  for (int r = 0; r < 8; ++r) { mrow[r] = -1e30f; lrow[r] = 0.f; }
#pragma unroll
  for (int n = 0; n < 4; ++n) o[n] = zero;

  const float scale = 0.125f;  // 1/sqrt(64)

  int pg[4], pe[4];
#pragma unroll
  for (int n = 0; n < 4; ++n) inv_k((n * 16 + nn) & 31, pg[n], pe[n]);

  const __bf16* kb = Kf + (size_t)bh * 128 * 2 * 512;  // per-iter: +4 tTiles
  const __bf16* vb = Vf + (size_t)bh * 64 * 4 * 512;   // per-iter: +2 tTiles

  stage16<512, 128>(Ks, kb, tid);
  stage16<512, 128>(Vs, vb, tid);
  for (int j = 0; j < S_ / 64; ++j) {
    if (j + 1 < S_ / 64) {
      stage16<512, 128>(Ks + ((j + 1) & 1) * 4096, kb + (size_t)(j + 1) * 4096, tid);
      stage16<512, 128>(Vs + ((j + 1) & 1) * 4096, vb + (size_t)(j + 1) * 4096, tid);
      async_wait<8>();  // 8 chunks/thread in flight for next stage
    } else {
      async_wait<0>();
    }
    __syncthreads();
    const __bf16* Kc = Ks + (j & 1) * 4096;
    const __bf16* Vc = Vs + (j & 1) * 4096;

    // scores: four 16x16 tiles, K=64 via 2 chained WMMAs each
    v8f sc[4];
#pragma unroll
    for (int n = 0; n < 4; ++n) {
      const v16bf b0 = load_frag(&Kc[(n * 2 + 0) * 512], lane);
      const v16bf b1 = load_frag(&Kc[(n * 2 + 1) * 512], lane);
      v8f zz = zero;
      zz = wmma_bf16(qa0, b0, zz);
      sc[n] = wmma_bf16(qa1, b1, zz);
    }

    // online softmax (row = r + 8*gc)
    float alpha[8];
#pragma unroll
    for (int r = 0; r < 8; ++r) {
      const float x0 = sc[0][r] * scale, x1 = sc[1][r] * scale;
      const float x2 = sc[2][r] * scale, x3 = sc[3][r] * scale;
      float mx = fmaxf(fmaxf(x0, x1), fmaxf(x2, x3));
#pragma unroll
      for (int off = 1; off < 16; off <<= 1)
        mx = fmaxf(mx, __shfl_xor(mx, off, 32));
      const float nm = fmaxf(mrow[r], mx);
      alpha[r] = __expf(mrow[r] - nm);
      const float p0 = __expf(x0 - nm), p1 = __expf(x1 - nm);
      const float p2 = __expf(x2 - nm), p3 = __expf(x3 - nm);
      sc[0][r] = p0; sc[1][r] = p1; sc[2][r] = p2; sc[3][r] = p3;
      float sm = (p0 + p1) + (p2 + p3);
#pragma unroll
      for (int off = 1; off < 16; off <<= 1)
        sm += __shfl_xor(sm, off, 32);
      lrow[r] = lrow[r] * alpha[r] + sm;
      mrow[r] = nm;
    }
#pragma unroll
    for (int n = 0; n < 4; ++n)
#pragma unroll
      for (int r = 0; r < 8; ++r) o[n][r] *= alpha[r];

    // probs C-layout -> 2 A-fragments (t 0..31 / 32..63) via LDS bounce
    __bf16* P = &Ps[wave][0];
#pragma unroll
    for (int n = 0; n < 4; ++n) {
      const int c = n >> 1;
#pragma unroll
      for (int r = 0; r < 8; ++r)
        P[c * 512 + ((r + 8 * gc) + 16 * pg[n]) * 16 + pe[n]] = (__bf16)sc[n][r];
    }
    const v16bf pa0 = load_frag(P, lane);        // same-wave LDS, in-order
    const v16bf pa1 = load_frag(P + 512, lane);

#pragma unroll
    for (int eT = 0; eT < 4; ++eT) {
      o[eT] = wmma_bf16(pa0, load_frag(&Vc[eT * 512], lane), o[eT]);
      o[eT] = wmma_bf16(pa1, load_frag(&Vc[(4 + eT) * 512], lane), o[eT]);
    }
    __syncthreads();
  }

  // normalize + store ctx into out_proj's A-fragment layout
  const int b = bh >> 4, h = bh & 15;
#pragma unroll
  for (int r = 0; r < 8; ++r) {
    const float inv = 1.0f / lrow[r];
    const int gm = b * S_ + s0 + wave * 16 + r + 8 * gc;
#pragma unroll
    for (int n = 0; n < 4; ++n) {
      const int col = h * DH_ + n * 16 + nn;
      int g, e;
      inv_k(col & 31, g, e);
      Cf[((size_t)(gm >> 4) * 32 + (col >> 5)) * 512 +
         ((gm & 15) + 16 * g) * 16 + e] = (__bf16)(o[n][r] * inv);
    }
  }
}

// ---------------------------------------------------------------------------
// Launcher
// ---------------------------------------------------------------------------
extern "C" void kernel_launch(void* const* d_in, const int* in_sizes, int n_in,
                              void* d_out, int out_size, void* d_ws, size_t ws_size,
                              hipStream_t stream) {
  const float* x  = (const float*)d_in[0];
  const float* Wq = (const float*)d_in[1];
  const float* bq = (const float*)d_in[2];
  const float* Wk = (const float*)d_in[3];
  const float* bk = (const float*)d_in[4];
  const float* Wv = (const float*)d_in[5];
  const float* bv = (const float*)d_in[6];
  const float* Wo = (const float*)d_in[7];
  const float* bo = (const float*)d_in[8];

  char* ws = (char*)d_ws;
  const size_t MB = 1024 * 1024;
  __bf16* Xs  = (__bf16*)(ws);            //  8 MB: x, A-frag tiled
  __bf16* Wqs = (__bf16*)(ws +  8 * MB);  //  2 MB each: B-frag tiled
  __bf16* Wks = (__bf16*)(ws + 10 * MB);
  __bf16* Wvs = (__bf16*)(ws + 12 * MB);
  __bf16* Wos = (__bf16*)(ws + 14 * MB);
  __bf16* Qf  = (__bf16*)(ws + 16 * MB);  //  8 MB each
  __bf16* Kf  = (__bf16*)(ws + 24 * MB);
  __bf16* Vf  = (__bf16*)(ws + 32 * MB);
  __bf16* Cf  = (__bf16*)(ws + 40 * MB);

  conv_X<<<(B_ * S_ * D_) / 256, 256, 0, stream>>>(Xs, x);
  WcArgs wc = {{Wq, Wk, Wv}, {Wqs, Wks, Wvs}};
  conv_Whead3<<<dim3((D_ * D_) / 256, 3), 256, 0, stream>>>(wc);
  conv_Wo<<<(D_ * D_) / 256, 256, 0, stream>>>(Wos, Wo);

  QKVArgs qa = {{Wqs, Wks, Wvs}, {bq, bk, bv}, {Qf, Kf, Vf}};
  gemm_qkv<<<dim3(B_ * S_ / 128, D_ / 128, 3), 256, 0, stream>>>(Xs, qa);

  attn<<<dim3(S_ / 64, B_ * H_), 128, 0, stream>>>(Qf, Kf, Vf, Cf);

  gemm_out<<<dim3(B_ * S_ / 128, D_ / 128), 256, 0, stream>>>(Cf, Wos, bo,
                                                              (float*)d_out);
}
